// CombinedGCN_59785944760956
// MI455X (gfx1250) — matrix operand order repaired and checked by reading, not compiled, not run on hardware
//
#include <hip/hip_runtime.h>
#include <hip/hip_bf16.h>

// ---------------------------------------------------------------------------
// CDNA5 GCN pipeline. Dense feature transforms use v_wmma_f32_16x16x32_bf16
// with K-unrolled loops and pre-packed (WMMA lane layout) B operands so both
// A and B come in via global_load_b128. Edge aggregation is an L2-resident
// atomic scatter-add (working set ~130MB < 192MB L2 on MI455X).
// ---------------------------------------------------------------------------

typedef __attribute__((ext_vector_type(16))) __bf16 v16bf;
typedef __attribute__((ext_vector_type(16))) unsigned short v16u;
typedef __attribute__((ext_vector_type(8)))  unsigned short v8u;
typedef __attribute__((ext_vector_type(8)))  float v8f;

#define GCN_N 50000
#define GCN_E 800000
#define GCN_G 64

__device__ __forceinline__ unsigned short f2bf(float f) {
    unsigned int u = __float_as_uint(f);
    u += 0x7FFFu + ((u >> 16) & 1u);          // round-to-nearest-even
    return (unsigned short)(u >> 16);
}

// ---------------- degree -----------------------------------------------------
__global__ void gcn_fill1_kernel(float* p, int n) {
    int i = blockIdx.x * blockDim.x + threadIdx.x;
    if (i < n) p[i] = 1.0f;                    // self-loop contributes 1
}
__global__ void gcn_degacc_kernel(const int* __restrict__ dst, float* deg, int e) {
    int i = blockIdx.x * blockDim.x + threadIdx.x;
    if (i < e) atomicAdd(&deg[dst[i]], 1.0f);
}
__global__ void gcn_rsqrt_kernel(float* p, int n) {
    int i = blockIdx.x * blockDim.x + threadIdx.x;
    if (i < n) p[i] = rsqrtf(p[i]);            // deg >= 1 always (self-loops)
}

// ---------------- bf16 conversion -------------------------------------------
// features: [rows x Kin] f32 -> [rows x Kp] bf16, zero-padded in K
__global__ void gcn_cvt_feat_kernel(const float* __restrict__ in, unsigned short* __restrict__ out,
                                    int rows, int Kin, int Kp) {
    int idx = blockIdx.x * blockDim.x + threadIdx.x;
    int total = rows * Kp;
    if (idx >= total) return;
    int r = idx / Kp, k = idx - r * Kp;
    out[idx] = (k < Kin) ? f2bf(in[(size_t)r * Kin + k]) : (unsigned short)0;
}
// weights: [Kin x Nn] f32 -> packed WMMA-B lane layout, zero-padded in K.
// Packed index: (((nt * Kp/32) + kt) * 32 + lane) * 16 + i
//   lane = 16*half + ln;  element i -> B[kt*32 + (i < 8 ? 8*half + i : 16 + 8*half + i - 8)][nt*16 + ln]
__global__ void gcn_pack_w_kernel(const float* __restrict__ W, unsigned short* __restrict__ Bp,
                                  int Kin, int Kp, int Nn) {
    int idx = blockIdx.x * blockDim.x + threadIdx.x;
    int total = (Nn >> 4) * (Kp >> 5) * 512;   // == Kp*Nn
    if (idx >= total) return;
    int i    = idx & 15;
    int lane = (idx >> 4) & 31;
    int tile = idx >> 9;
    int ktiles = Kp >> 5;
    int nt = tile / ktiles, kt = tile - nt * ktiles;
    int half = lane >> 4, ln = lane & 15;
    int k = kt * 32 + (i < 8 ? 8 * half + i : 16 + 8 * half + (i - 8));
    int n = nt * 16 + ln;
    Bp[idx] = (k < Kin) ? f2bf(W[(size_t)k * Nn + n]) : (unsigned short)0;
}

// ---------------- WMMA GEMM: C[MxNn] = A[MxK](bf16) * Bpacked(bf16) ---------
// One wave per 16x16 output tile; K templated so the loop fully unrolls.
template<int K>
__global__ void __launch_bounds__(32)
gcn_wmma_gemm_kernel(const unsigned short* __restrict__ A,
                     const unsigned short* __restrict__ Bp,
                     float* __restrict__ C, int Nn) {
    const int lane = threadIdx.x & 31;
    const int half = lane >> 4;      // 0: lanes 0-15, 1: lanes 16-31
    const int ln   = lane & 15;
    const int m0   = blockIdx.x * 16;
    const int n0t  = blockIdx.y;     // n-tile index
    const int row  = m0 + ln;        // A-matrix row held by this lane

    const unsigned short* arow = A + (size_t)row * K + 8 * half;
    const unsigned short* bbase = Bp + ((size_t)(n0t * (K / 32)) * 32 + lane) * 16;

    v8f acc = {};
#pragma unroll
    for (int kt = 0; kt < K / 32; ++kt) {
        // A 16x32 bf16 tile (ISA 7.12.2): lane holds row m, K = {8h..8h+7, 16+8h..+7}
        v8u alo = *(const v8u*)(arow + kt * 32);
        v8u ahi = *(const v8u*)(arow + kt * 32 + 16);
        // B 32x16 bf16 tile, pre-packed: 32 contiguous bytes per lane
        const unsigned short* bp = bbase + (size_t)kt * 512;
        v8u blo = *(const v8u*)(bp);
        v8u bhi = *(const v8u*)(bp + 8);
        v16u av, bv;
#pragma unroll
        for (int i = 0; i < 8; ++i) {
            av[i] = alo[i]; av[8 + i] = ahi[i];
            bv[i] = blo[i]; bv[8 + i] = bhi[i];
        }
        v16bf a = __builtin_bit_cast(v16bf, av);
        v16bf b = __builtin_bit_cast(v16bf, bv);
        acc = __builtin_amdgcn_wmma_f32_16x16x32_bf16(
            /*neg_a=*/false, a, /*neg_b=*/false, b,
            /*c_mod=*/(short)0, acc, /*reuse_a=*/false, /*reuse_b=*/false);
    }
    // D layout: VGPR r -> row m0 + r + 8*half, col n0 + ln
#pragma unroll
    for (int r = 0; r < 8; ++r)
        C[(size_t)(m0 + r + 8 * half) * Nn + n0t * 16 + ln] = acc[r];
}

// ---------------- aggregation -----------------------------------------------
// post[i][j] = bias[j] + dinv[i]^2 * pre[i][j]   (bias + self-loop term)
__global__ void gcn_initself_kernel(const float* __restrict__ pre, const float* __restrict__ bias,
                                    const float* __restrict__ dinv, float* __restrict__ post,
                                    int rows, int logF) {
    int idx = blockIdx.x * blockDim.x + threadIdx.x;
    int F = 1 << logF;
    if (idx >= (rows << logF)) return;
    int i = idx >> logF, j = idx & (F - 1);
    float w = dinv[i];
    post[idx] = bias[j] + w * w * pre[idx];
}
// scatter real edges: 32 lanes per edge, lane strides columns (coalesced rows)
template<int F>
__global__ void gcn_edge_scatter_kernel(const int* __restrict__ src, const int* __restrict__ dst,
                                        const float* __restrict__ dinv,
                                        const float* __restrict__ pre, float* __restrict__ post,
                                        int e) {
    int t = blockIdx.x * blockDim.x + threadIdx.x;
    int edge = t >> 5;
    int lane = t & 31;
    if (edge >= e) return;
    int s = src[edge], d = dst[edge];
    float w = dinv[s] * dinv[d];
    const float* ps = pre + (size_t)s * F;
    float* pd = post + (size_t)d * F;
#pragma unroll
    for (int j = lane; j < F; j += 32)
        atomicAdd(&pd[j], w * ps[j]);
}
__global__ void gcn_relu_kernel(float* h, int total) {
    int i = blockIdx.x * blockDim.x + threadIdx.x;
    if (i < total) h[i] = fmaxf(h[i], 0.0f);
}

// ---------------- batchnorm --------------------------------------------------
__global__ void gcn_bn_stats_kernel(const float* __restrict__ h, int rows, int F,
                                    float* __restrict__ mu, float* __restrict__ rinv) {
    int c = blockIdx.x;
    __shared__ float ss[256];
    __shared__ float s2[256];
    float a = 0.0f, b = 0.0f;
    for (int i = threadIdx.x; i < rows; i += blockDim.x) {
        float v = h[(size_t)i * F + c];
        a += v; b += v * v;
    }
    ss[threadIdx.x] = a; s2[threadIdx.x] = b;
    __syncthreads();
    for (int o = 128; o > 0; o >>= 1) {
        if ((int)threadIdx.x < o) { ss[threadIdx.x] += ss[threadIdx.x + o]; s2[threadIdx.x] += s2[threadIdx.x + o]; }
        __syncthreads();
    }
    if (threadIdx.x == 0) {
        float m = ss[0] / (float)rows;
        float var = s2[0] / (float)rows - m * m;
        mu[c] = m;
        rinv[c] = rsqrtf(var + 1e-5f);
    }
}
__global__ void gcn_bn_apply_kernel(float* h, const float* __restrict__ mu, const float* __restrict__ rinv,
                                    const float* __restrict__ gamma, const float* __restrict__ beta,
                                    int rows, int logF) {
    int idx = blockIdx.x * blockDim.x + threadIdx.x;
    int F = 1 << logF;
    if (idx >= (rows << logF)) return;
    int c = idx & (F - 1);
    h[idx] = (h[idx] - mu[c]) * rinv[c] * gamma[c] + beta[c];
}

// ---------------- pooling + FC -----------------------------------------------
__global__ void gcn_pool_zero_kernel(float* psum, float* pcnt) {
    int i = blockIdx.x * blockDim.x + threadIdx.x;
    if (i < GCN_G * 64) psum[i] = 0.0f;
    if (i < GCN_G) pcnt[i] = 0.0f;
}
__global__ void gcn_pool_accum_kernel(const float* __restrict__ h, const int* __restrict__ batch,
                                      float* psum, float* pcnt, int rows) {
    int idx = blockIdx.x * blockDim.x + threadIdx.x;
    if (idx >= rows * 64) return;
    int i = idx >> 6, c = idx & 63;
    int g = batch[i];
    atomicAdd(&psum[g * 64 + c], h[(size_t)i * 64 + c]);
    if (c == 0) atomicAdd(&pcnt[g], 1.0f);
}
__global__ void gcn_fc_kernel(const float* __restrict__ psum, const float* __restrict__ pcnt,
                              const float* __restrict__ Wfc, const float* __restrict__ bfc,
                              float* __restrict__ out) {
    int t = blockIdx.x * blockDim.x + threadIdx.x;
    if (t >= GCN_G * 6) return;
    int g = t / 6, k = t - g * 6;
    float inv = 1.0f / fmaxf(pcnt[g], 1.0f);
    float acc = bfc[k];
#pragma unroll
    for (int c = 0; c < 64; ++c)
        acc += psum[g * 64 + c] * inv * Wfc[c * 6 + k];
    out[t] = acc;
}

// ---------------------------------------------------------------------------
template<int Kp, int Nn, int LOGF>
static void run_gcn_layer(const float* hin, int Kin,
                          const float* W, const float* bias,
                          const int* src, const int* dst, const float* dinv,
                          unsigned short* hbf, unsigned short* wbf,
                          float* pre, float* post, hipStream_t stream) {
    const int N = GCN_N, E = GCN_E;
    int tot = N * Kp;
    gcn_cvt_feat_kernel<<<(tot + 255) / 256, 256, 0, stream>>>(hin, hbf, N, Kin, Kp);
    int wt = Kp * Nn;
    gcn_pack_w_kernel<<<(wt + 255) / 256, 256, 0, stream>>>(W, wbf, Kin, Kp, Nn);
    dim3 grid(N / 16, Nn / 16);
    gcn_wmma_gemm_kernel<Kp><<<grid, 32, 0, stream>>>(hbf, wbf, pre, Nn);
    tot = N * Nn;
    gcn_initself_kernel<<<(tot + 255) / 256, 256, 0, stream>>>(pre, bias, dinv, post, N, LOGF);
    long long et = (long long)E * 32;
    gcn_edge_scatter_kernel<Nn><<<(int)((et + 255) / 256), 256, 0, stream>>>(src, dst, dinv, pre, post, E);
    gcn_relu_kernel<<<(tot + 255) / 256, 256, 0, stream>>>(post, tot);
}

extern "C" void kernel_launch(void* const* d_in, const int* in_sizes, int n_in,
                              void* d_out, int out_size, void* d_ws, size_t ws_size,
                              hipStream_t stream) {
    const float* x    = (const float*)d_in[0];
    const int*   edge = (const int*)d_in[1];
    const int*   batch= (const int*)d_in[2];
    const float* W1 = (const float*)d_in[3];  const float* b1  = (const float*)d_in[4];
    const float* W2 = (const float*)d_in[5];  const float* b2  = (const float*)d_in[6];
    const float* g1 = (const float*)d_in[7];  const float* be1 = (const float*)d_in[8];
    const float* W3 = (const float*)d_in[9];  const float* b3  = (const float*)d_in[10];
    const float* g2 = (const float*)d_in[11]; const float* be2 = (const float*)d_in[12];
    const float* Wfc= (const float*)d_in[13]; const float* bfc = (const float*)d_in[14];

    const int N = GCN_N, E = GCN_E;
    const int* src = edge;
    const int* dst = edge + E;

    // workspace carve-out (~130 MB total; L2-resident on MI455X)
    char* ws = (char*)d_ws;
    size_t off = 0;
    auto alloc = [&](size_t bytes) { size_t o = off; off = (off + bytes + 255) & ~(size_t)255; return o; };
    float*          dinv = (float*)(ws + alloc((size_t)N * 4));
    float*          pre  = (float*)(ws + alloc((size_t)N * 256 * 4));
    float*          post = (float*)(ws + alloc((size_t)N * 256 * 4));
    unsigned short* hbf  = (unsigned short*)(ws + alloc((size_t)N * 256 * 2));
    unsigned short* wbf  = (unsigned short*)(ws + alloc((size_t)256 * 256 * 2));
    float*          mu   = (float*)(ws + alloc(256 * 4));
    float*          rinv = (float*)(ws + alloc(256 * 4));
    float*          psum = (float*)(ws + alloc((size_t)GCN_G * 64 * 4));
    float*          pcnt = (float*)(ws + alloc((size_t)GCN_G * 4));
    (void)ws_size; (void)in_sizes; (void)n_in; (void)out_size;

    // degree (self-loops included) -> deg^{-1/2}
    gcn_fill1_kernel<<<(N + 255) / 256, 256, 0, stream>>>(dinv, N);
    gcn_degacc_kernel<<<(E + 255) / 256, 256, 0, stream>>>(dst, dinv, E);
    gcn_rsqrt_kernel<<<(N + 255) / 256, 256, 0, stream>>>(dinv, N);

    // layer 1: [N,5] -> [N,256], relu
    run_gcn_layer<32, 256, 8>(x, 5, W1, b1, src, dst, dinv, hbf, wbf, pre, post, stream);

    // layer 2: [N,256] -> [N,128], relu, bn1
    run_gcn_layer<256, 128, 7>(post, 256, W2, b2, src, dst, dinv, hbf, wbf, pre, post, stream);
    gcn_bn_stats_kernel<<<128, 256, 0, stream>>>(post, N, 128, mu, rinv);
    gcn_bn_apply_kernel<<<(N * 128 + 255) / 256, 256, 0, stream>>>(post, mu, rinv, g1, be1, N, 7);

    // layer 3: [N,128] -> [N,64], relu, bn2
    run_gcn_layer<128, 64, 6>(post, 128, W3, b3, src, dst, dinv, hbf, wbf, pre, post, stream);
    gcn_bn_stats_kernel<<<64, 256, 0, stream>>>(post, N, 64, mu, rinv);
    gcn_bn_apply_kernel<<<(N * 64 + 255) / 256, 256, 0, stream>>>(post, mu, rinv, g2, be2, N, 6);

    // global mean pool + FC
    gcn_pool_zero_kernel<<<(GCN_G * 64 + 255) / 256, 256, 0, stream>>>(psum, pcnt);
    gcn_pool_accum_kernel<<<(N * 64 + 255) / 256, 256, 0, stream>>>(post, batch, psum, pcnt, N);
    gcn_fc_kernel<<<(GCN_G * 6 + 255) / 256, 256, 0, stream>>>(psum, pcnt, Wfc, bfc, (float*)d_out);
}
